// Faster_RCNN_85598698209578
// MI455X (gfx1250) — compile-verified, matrix operand order checked by reading, not run on hardware
//
#include <hip/hip_runtime.h>

// ---------------------------------------------------------------------------
// Types for CDNA5 WMMA (gfx1250, wave32)
// ---------------------------------------------------------------------------
typedef __attribute__((ext_vector_type(16))) __bf16    v16bf;
typedef __attribute__((ext_vector_type(8)))  float     v8f;
typedef __attribute__((ext_vector_type(4)))  unsigned  u32x4;

#define DEV __device__ __forceinline__

struct U32x8 { u32x4 lo, hi; };

static DEV v16bf tov16(u32x4 lo, u32x4 hi) {
  U32x8 t{lo, hi};
  return __builtin_bit_cast(v16bf, t);
}

// round-to-nearest-even fp32 -> bf16
static DEV unsigned short f2bf(float f) {
  unsigned u = __builtin_bit_cast(unsigned, f);
  unsigned r = (u + 0x7fffu + ((u >> 16) & 1u)) >> 16;
  return (unsigned short)r;
}

// ---------------------------------------------------------------------------
// Problem constants
// ---------------------------------------------------------------------------
#define BB   8
#define HH   64
#define WW   64
#define CI   512
#define CO   256
#define NTAP 9
#define NHEAD 48   // 36 reg + 9 cls + 3 pad
#define OUTC 45

// workspace layout (bytes)
#define WS_DATA_OFF  0
#define WS_DATA_SZ   (BB*HH*WW*CI*2)            // 33,554,432
#define WS_WB_OFF    (WS_DATA_OFF + WS_DATA_SZ)
#define WS_WB_SZ     (NTAP*CO*CI*2)             //  2,359,296
#define WS_WH_OFF    (WS_WB_OFF + WS_WB_SZ)
#define WS_WH_SZ     (NHEAD*CO*2)

// A-slab staging geometry: 64 rows x 64 channels (bf16), row stride 72 halves
#define SA_STRIDE_H  72          // halves (144 B = 9*16, aligned + conflict-free)
#define SA_BUF_H     (64 * SA_STRIDE_H)

// ---------------------------------------------------------------------------
// Prep kernels: fp32 -> bf16 with layout changes
// ---------------------------------------------------------------------------
__global__ void cvt_data_k(const float* __restrict__ src,
                           unsigned short* __restrict__ dst, int n4) {
  int i = blockIdx.x * blockDim.x + threadIdx.x;
  if (i >= n4) return;
  float4 v = ((const float4*)src)[i];
  ushort4 o;
  o.x = f2bf(v.x); o.y = f2bf(v.y); o.z = f2bf(v.z); o.w = f2bf(v.w);
  ((ushort4*)dst)[i] = o;
}

// w_base HWIO [ky][kx][c][n] -> bf16 [tap][n][c]
__global__ void cvt_wbase_k(const float* __restrict__ w,
                            unsigned short* __restrict__ dst) {
  int i = blockIdx.x * blockDim.x + threadIdx.x;   // NTAP*CO*CI
  if (i >= NTAP*CO*CI) return;
  int c = i & (CI - 1);
  int n = (i >> 9) & (CO - 1);
  int t = i >> 17;
  dst[i] = f2bf(w[(t*CI + c)*CO + n]);
}

// combined head weights: bf16 [n][k], n<36: w_reg[k][n], 36<=n<45: w_cls[k][n-36]
__global__ void cvt_whead_k(const float* __restrict__ wcls,
                            const float* __restrict__ wreg,
                            unsigned short* __restrict__ dst) {
  int i = blockIdx.x * blockDim.x + threadIdx.x;   // NHEAD*CO
  if (i >= NHEAD*CO) return;
  int k = i & (CO - 1);
  int n = i >> 8;
  float v = 0.f;
  if (n < 36)      v = wreg[k*36 + n];
  else if (n < 45) v = wcls[k*9 + (n - 36)];
  dst[i] = f2bf(v);
}

// ---------------------------------------------------------------------------
// Async staging of one 64-row x 64-channel A slab into an LDS buffer.
// Each thread covers 32 B (two b128 transfers). OOB lanes write zeros.
// ---------------------------------------------------------------------------
static DEV void stage_step(unsigned short* sbuf,
                           const unsigned short* __restrict__ dataBF,
                           int b, int h, int step, int srow, int sseg) {
  const int t  = step >> 3;
  const int kc = (step & 7) * 64;
  const int dy = t / 3 - 1, dx = t - (t / 3) * 3 - 1;
  const int h2 = h + dy;
  const int w2 = srow + dx;
  const bool ok = ((unsigned)h2 < (unsigned)HH) && ((unsigned)w2 < (unsigned)WW);
  char* ldst = (char*)sbuf + srow * (SA_STRIDE_H * 2) + sseg * 32;
  if (ok) {
    const char* g = (const char*)(dataBF +
        (size_t)(((b << 6) + h2) * 64 + w2) * CI + kc + sseg * 16);
    // low 32 bits of a generic LDS pointer == LDS byte offset (flat aperture)
    unsigned           la = (unsigned)(size_t)ldst;
    unsigned long long ga = (unsigned long long)(size_t)g;
    asm volatile("global_load_async_to_lds_b128 %0, %1, off"
                 :: "v"(la), "v"(ga) : "memory");
    asm volatile("global_load_async_to_lds_b128 %0, %1, off offset:16"
                 :: "v"(la), "v"(ga) : "memory");
  } else {
    const u32x4 z = {0u, 0u, 0u, 0u};
    *(u32x4*)(ldst)      = z;
    *(u32x4*)(ldst + 16) = z;
  }
}

// ---------------------------------------------------------------------------
// Fused RPN kernel: one WG per (b,h) row. Tile M=64(w) x N=256, K=9*512.
// 8 waves; wave owns 32 output channels (2 N-frags) x 4 M-frags.
// Double-buffered async global->LDS staging of the A operand.
// ---------------------------------------------------------------------------
__launch_bounds__(256)
__global__ void rpn_fused_k(const unsigned short* __restrict__ dataBF,
                            const unsigned short* __restrict__ wBaseBF,
                            const unsigned short* __restrict__ wHeadBF,
                            const float* __restrict__ b_base,
                            const float* __restrict__ b_cls,
                            const float* __restrict__ b_reg,
                            float* __restrict__ out) {
  __shared__ unsigned short sA2[2 * SA_BUF_H];  // 18.0 KB (double buffer)
  __shared__ unsigned short sFeat[64 * 264];    // 33.8 KB
  __shared__ float          sRaw[64 * 48];      // 12.3 KB

  const int wg   = blockIdx.x;        // 0..511
  const int b    = wg >> 6;
  const int h    = wg & 63;
  const int tid  = threadIdx.x;
  const int wid  = tid >> 5;
  const int lane = tid & 31;
  const int ln16 = lane & 15;
  const int hi16 = lane >> 4;         // 0/1
  const int kb   = hi16 * 8;          // per-lane K base (halves) per ISA layout
  const int n0   = wid * 32;          // this wave's output-channel base

  const v8f zacc = {0.f,0.f,0.f,0.f,0.f,0.f,0.f,0.f};
  v8f acc[4][2];
  #pragma unroll
  for (int m = 0; m < 4; ++m) { acc[m][0] = zacc; acc[m][1] = zacc; }

  // staging role: thread -> (row = pixel w, 32B segment of 64 channels)
  const int srow = tid >> 2;          // 0..63
  const int sseg = tid & 3;           // 0..3

  const int NSTEP = NTAP * 8;         // 72 steps of 64 channels

  // prologue: stage step 0 into buffer 0
  stage_step(sA2, dataBF, b, h, 0, srow, sseg);
  asm volatile("s_wait_asynccnt 0x0" ::: "memory");
  __syncthreads();

  for (int s = 0; s < NSTEP; ++s) {
    const int cur = s & 1;
    unsigned short* bufC = sA2 + cur * SA_BUF_H;

    // kick off DMA for the next slab into the other buffer (overlaps WMMA)
    if (s + 1 < NSTEP)
      stage_step(sA2 + (1 - cur) * SA_BUF_H, dataBF, b, h, s + 1, srow, sseg);

    const int t  = s >> 3;
    const int kc = (s & 7) * 64;
    const unsigned short* wTap = wBaseBF + (size_t)t * CO * CI;

    #pragma unroll
    for (int sub = 0; sub < 2; ++sub) {
      const int kcs = kc + sub * 32;
      // A fragments (16x32 bf16) from LDS: two b128 per frag
      u32x4 aLo[4], aHi[4];
      #pragma unroll
      for (int m = 0; m < 4; ++m) {
        const char* base = (const char*)bufC +
            (m * 16 + ln16) * (SA_STRIDE_H * 2) + sub * 64 + kb * 2;
        aLo[m] = *(const u32x4*)(base);
        aHi[m] = *(const u32x4*)(base + 32);
      }
      // B fragments (32x16 bf16) straight from global ([tap][n][k] layout)
      v16bf Bf[2];
      #pragma unroll
      for (int f = 0; f < 2; ++f) {
        const unsigned short* bp =
            wTap + (size_t)(n0 + f * 16 + ln16) * CI + kcs + kb;
        Bf[f] = tov16(*(const u32x4*)bp, *(const u32x4*)(bp + 16));
      }
      #pragma unroll
      for (int m = 0; m < 4; ++m) {
        v16bf Af = tov16(aLo[m], aHi[m]);
        #pragma unroll
        for (int f = 0; f < 2; ++f)
          acc[m][f] = __builtin_amdgcn_wmma_f32_16x16x32_bf16(
              false, Af, false, Bf[f], (short)0, acc[m][f], false, false);
      }
    }

    // next slab's DMA must have landed before anyone reads it
    if (s + 1 < NSTEP)
      asm volatile("s_wait_asynccnt 0x0" ::: "memory");
    __syncthreads();
  }

  // ---- epilogue 1: bias + ReLU -> bf16 feature tile in LDS ----
  float bbias[2];
  #pragma unroll
  for (int f = 0; f < 2; ++f) bbias[f] = b_base[n0 + f * 16 + ln16];

  #pragma unroll
  for (int m = 0; m < 4; ++m)
    #pragma unroll
    for (int f = 0; f < 2; ++f) {
      const int col = n0 + f * 16 + ln16;
      #pragma unroll
      for (int v = 0; v < 8; ++v) {
        const int row = m * 16 + hi16 * 8 + v;
        float x = acc[m][f][v] + bbias[f];
        x = x > 0.f ? x : 0.f;
        sFeat[row * 264 + col] = f2bf(x);
      }
    }
  __syncthreads();

  // ---- head GEMM: 64 x 48, K=256; waves 0-3: j 0..31, waves 4-7: j 32..47
  const int mf2 = wid & 3;
  const int nfr = (wid < 4) ? 2 : 1;
  const int jb0 = (wid < 4) ? 0 : 32;
  v8f acc2[2]; acc2[0] = zacc; acc2[1] = zacc;

  for (int kc = 0; kc < CO; kc += 32) {
    const char* abase =
        (const char*)sFeat + (mf2 * 16 + ln16) * 528 + (kc + kb) * 2;
    v16bf Af = tov16(*(const u32x4*)abase, *(const u32x4*)(abase + 32));
    #pragma unroll
    for (int f = 0; f < 2; ++f) {
      if (f < nfr) {   // wave-uniform predicate: EXEC stays all-ones per wave
        const unsigned short* bp =
            wHeadBF + (size_t)(jb0 + f * 16 + ln16) * CO + kc + kb;
        v16bf Bf = tov16(*(const u32x4*)bp, *(const u32x4*)(bp + 16));
        acc2[f] = __builtin_amdgcn_wmma_f32_16x16x32_bf16(
            false, Af, false, Bf, (short)0, acc2[f], false, false);
      }
    }
  }

  // ---- raw scores (+bias) to LDS ----
  #pragma unroll
  for (int f = 0; f < 2; ++f) {
    if (f < nfr) {
      const int j = jb0 + f * 16 + ln16;
      const float bj = (j < 36) ? b_reg[j] : (j < 45 ? b_cls[j - 36] : 0.f);
      #pragma unroll
      for (int v = 0; v < 8; ++v) {
        const int row = mf2 * 16 + hi16 * 8 + v;
        sRaw[row * 48 + j] = acc2[f][v] + bj;
      }
    }
  }
  __syncthreads();

  // ---- epilogue 2: sigmoid / selection mask / dense pack [36 box, 9 obj]
  float* outBase = out + (size_t)((b * 64 + h) * 64) * OUTC;
  for (int idx = tid; idx < 64 * OUTC; idx += 256) {
    const int m = idx / OUTC;
    const int j = idx - m * OUTC;
    float val;
    if (j >= 36) {
      val = 1.f / (1.f + __expf(-sRaw[m * 48 + j]));      // objectness
    } else {
      const int a  = j >> 2;
      const float o  = 1.f / (1.f + __expf(-sRaw[m * 48 + 36 + a]));
      const float bw = sRaw[m * 48 + a * 4 + 2];
      const float bh = sRaw[m * 48 + a * 4 + 3];
      const bool keep = (o > 0.7f) && (bw > 10.f) && (bh > 10.f);
      val = keep ? sRaw[m * 48 + j] : 0.f;
    }
    outBase[m * OUTC + j] = val;
  }
}

// ---------------------------------------------------------------------------
// Launch
// ---------------------------------------------------------------------------
extern "C" void kernel_launch(void* const* d_in, const int* in_sizes, int n_in,
                              void* d_out, int out_size, void* d_ws, size_t ws_size,
                              hipStream_t stream) {
  const float* data  = (const float*)d_in[0];
  const float* wbase = (const float*)d_in[1];
  const float* bbase = (const float*)d_in[2];
  const float* wcls  = (const float*)d_in[3];
  const float* bcls  = (const float*)d_in[4];
  const float* wreg  = (const float*)d_in[5];
  const float* breg  = (const float*)d_in[6];
  float* out = (float*)d_out;

  unsigned short* dataBF  = (unsigned short*)((char*)d_ws + WS_DATA_OFF);
  unsigned short* wBaseBF = (unsigned short*)((char*)d_ws + WS_WB_OFF);
  unsigned short* wHeadBF = (unsigned short*)((char*)d_ws + WS_WH_OFF);

  const int nData4 = (BB * HH * WW * CI) / 4;
  cvt_data_k<<<(nData4 + 255) / 256, 256, 0, stream>>>(data, dataBF, nData4);

  const int nWB = NTAP * CO * CI;
  cvt_wbase_k<<<(nWB + 255) / 256, 256, 0, stream>>>(wbase, wBaseBF);

  const int nWH = NHEAD * CO;
  cvt_whead_k<<<(nWH + 255) / 256, 256, 0, stream>>>(wcls, wreg, wHeadBF);

  rpn_fused_k<<<BB * HH, 256, 0, stream>>>(dataBF, wBaseBF, wHeadBF,
                                           bbase, bcls, breg, out);
}
